// LSTM_13056700580364
// MI455X (gfx1250) — compile-verified, matrix operand order checked by reading, not run on hardware
//
#include <hip/hip_runtime.h>

// ---------------------------------------------------------------------------
// CDNA5 / gfx1250 social-LSTM rollout.
// Gate GEMMs run on v_wmma_f32_16x16x32_f16 (fp16 in, fp32 accumulate).
// Weights are pre-swizzled into the exact WMMA B-fragment lane layout so the
// hot loop loads one fragment per lane with a single 32B load.
//
// `lz` (always 0) is laundered through an empty asm each time step so
// loop-invariant weight addresses cannot be hoisted by LICM (round-1 asm
// showed full hoist + scratch spill of ~6KB/lane without it).
//
// Fragment loads are batched per gate-tile (6 loads -> 6 WMMAs) so only one
// load-wait is exposed per gate instead of one per WMMA (round-2 asm showed
// load->wait->wmma fully serialized).
// ---------------------------------------------------------------------------

typedef __attribute__((ext_vector_type(16))) _Float16 v16h;
typedef __attribute__((ext_vector_type(8)))  _Float16 v8h;
typedef __attribute__((ext_vector_type(8)))  float    v8f;

#define WMMA_F16(A, B, C) \
  __builtin_amdgcn_wmma_f32_16x16x32_f16(false, (A), false, (B), (short)0, (C), false, false)

static constexpr int N_AGENTS = 65536;
static constexpr int HID      = 128;

// ---- workspace layout (bytes) ---------------------------------------------
static constexpr size_t OFF_E1_IH  = 0;                            // 512x64  f16
static constexpr size_t OFF_E1_HH  = OFF_E1_IH  + 512*64*2;        // 512x128 f16
static constexpr size_t OFF_E2_IHX = OFF_E1_HH  + 512*128*2;       // 512x64  f16
static constexpr size_t OFF_E2_IHH = OFF_E2_IHX + 512*64*2;        // 512x128 f16
static constexpr size_t OFF_E2_HH  = OFF_E2_IHH + 512*128*2;       // 512x128 f16
static constexpr size_t OFF_D_IH   = OFF_E2_HH  + 512*128*2;       // 512x64  f16
static constexpr size_t OFF_D_HH   = OFF_D_IH   + 512*64*2;        // 512x128 f16
static constexpr size_t OFF_H1     = OFF_D_HH   + 512*128*2;       // N*128 f16
static constexpr size_t OFF_H2     = OFF_H1 + (size_t)N_AGENTS*HID*2;
static constexpr size_t OFF_C2     = OFF_H2 + (size_t)N_AGENTS*HID*2; // N*128 f32

// ---------------------------------------------------------------------------
// helpers
// ---------------------------------------------------------------------------
__device__ __forceinline__ float sigf(float x) {
  return 1.0f / (1.0f + __expf(-x));
}
__device__ __forceinline__ float tanh_f(float x) {
  float xc = fminf(fmaxf(x, -15.0f), 15.0f);
  float e  = __expf(-2.0f * xc);
  return (1.0f - e) / (1.0f + e);
}

// One 32B swizzled B-fragment load (lane-contiguous).
__device__ __forceinline__ v16h ldfrag(const _Float16* __restrict__ base,
                                       int fragIdx, int lane, int lz) {
  return *(const v16h*)(base + (fragIdx * 32 + lane) * 16 + lz);
}

// Build A-fragments (4x v16h covering K=128) from a row-major [128]-feature
// f16 row (LDS h-tile row or global h row).  Fragment element e of lane
// (a, hi) holds feature k = 32q + (e/8)*16 + hi*8 + (e%8).
__device__ __forceinline__ void load_frags_row(const _Float16* row, int hi, v16h* fr) {
#pragma unroll
  for (int q = 0; q < 4; ++q) {
    v8h lo = *(const v8h*)(row + q * 32 + hi * 8);
    v8h hi8 = *(const v8h*)(row + q * 32 + 16 + hi * 8);
    fr[q] = __builtin_shufflevector(lo, hi8, 0,1,2,3,4,5,6,7,8,9,10,11,12,13,14,15);
  }
}

// InputEmbedding: relu((vel*4) @ W.T + b), built directly in A-fragment layout
// (2 fragments, K = 64) with VALU -- cheaper than a K=2 WMMA.
__device__ __forceinline__ void emb_frags(float vx, float vy,
                                          const float* s_embw, const float* s_embb,
                                          int hi, v16h* xfrag, int lz) {
#pragma unroll
  for (int q = 0; q < 2; ++q) {
    v16h f;
#pragma unroll
    for (int e = 0; e < 16; ++e) {
      int k = q * 32 + ((e >> 3) << 4) + hi * 8 + (e & 7);
      float v = 4.0f * (vx * s_embw[2 * k + lz] + vy * s_embw[2 * k + 1 + lz]) +
                s_embb[k + lz];
      f[e] = (_Float16)(v > 0.0f ? v : 0.0f);
    }
    xfrag[q] = f;
  }
}

// One LSTM cell step for a 16-agent tile.
//   gates = x@Wih^T (+ x2@Wih2^T) + h@Whh^T + bias   (torch order i,f,g,o)
//   c' = sig(f)*c + sig(i)*tanh(g);  h' = sig(o)*tanh(c')  -> stored to s_hw.
template <bool HAS2>
__device__ __forceinline__ void lstm_step(
    const _Float16* __restrict__ wih,   // swizzled, KC=2
    const _Float16* __restrict__ whh,   // swizzled, KC=4
    const _Float16* __restrict__ wih2,  // swizzled, KC=4 (enc2 h1-part) or null
    const v16h* xfrag, const v16h* hfrag, const v16h* x2frag,
    const float* __restrict__ s_bias, int lane, v8f* c,
    _Float16* __restrict__ s_hw, int lz) {
  const int l15 = lane & 15;
  const int hi  = lane >> 4;
#pragma unroll
  for (int nt = 0; nt < 8; ++nt) {
    v8f acc[4];
#pragma unroll
    for (int g = 0; g < 4; ++g) {
      const int ot = g * 8 + nt;
      float b = s_bias[g * 128 + nt * 16 + l15 + lz];
      v8f a = (v8f){b, b, b, b, b, b, b, b};
      if (!HAS2) {
        // batch all 6 fragment loads (one clause / one wait), then 6 WMMAs
        v16h b0 = ldfrag(wih, ot * 2 + 0, lane, lz);
        v16h b1 = ldfrag(wih, ot * 2 + 1, lane, lz);
        v16h h0 = ldfrag(whh, ot * 4 + 0, lane, lz);
        v16h h1 = ldfrag(whh, ot * 4 + 1, lane, lz);
        v16h h2 = ldfrag(whh, ot * 4 + 2, lane, lz);
        v16h h3 = ldfrag(whh, ot * 4 + 3, lane, lz);
        a = WMMA_F16(xfrag[0], b0, a);
        a = WMMA_F16(xfrag[1], b1, a);
        a = WMMA_F16(hfrag[0], h0, a);
        a = WMMA_F16(hfrag[1], h1, a);
        a = WMMA_F16(hfrag[2], h2, a);
        a = WMMA_F16(hfrag[3], h3, a);
      } else {
        // group 1: emb part + h1 part (6 loads -> 6 WMMAs)
        v16h b0 = ldfrag(wih, ot * 2 + 0, lane, lz);
        v16h b1 = ldfrag(wih, ot * 2 + 1, lane, lz);
        v16h p0 = ldfrag(wih2, ot * 4 + 0, lane, lz);
        v16h p1 = ldfrag(wih2, ot * 4 + 1, lane, lz);
        v16h p2 = ldfrag(wih2, ot * 4 + 2, lane, lz);
        v16h p3 = ldfrag(wih2, ot * 4 + 3, lane, lz);
        a = WMMA_F16(xfrag[0], b0, a);
        a = WMMA_F16(xfrag[1], b1, a);
        a = WMMA_F16(x2frag[0], p0, a);
        a = WMMA_F16(x2frag[1], p1, a);
        a = WMMA_F16(x2frag[2], p2, a);
        a = WMMA_F16(x2frag[3], p3, a);
        // group 2: recurrent part (4 loads -> 4 WMMAs)
        v16h h0 = ldfrag(whh, ot * 4 + 0, lane, lz);
        v16h h1 = ldfrag(whh, ot * 4 + 1, lane, lz);
        v16h h2 = ldfrag(whh, ot * 4 + 2, lane, lz);
        v16h h3 = ldfrag(whh, ot * 4 + 3, lane, lz);
        a = WMMA_F16(hfrag[0], h0, a);
        a = WMMA_F16(hfrag[1], h1, a);
        a = WMMA_F16(hfrag[2], h2, a);
        a = WMMA_F16(hfrag[3], h3, a);
      }
      acc[g] = a;
    }
    // elementwise LSTM update in D-layout: lane holds neuron nt*16+l15,
    // VGPR r holds agent m = r + hi*8.
#pragma unroll
    for (int r = 0; r < 8; ++r) {
      float ig = sigf(acc[0][r]);
      float fg = sigf(acc[1][r]);
      float gg = tanh_f(acc[2][r]);
      float og = sigf(acc[3][r]);
      float cn = fg * c[nt][r] + ig * gg;
      c[nt][r] = cn;
      float hn = og * tanh_f(cn);
      int m = r + hi * 8;
      s_hw[m * 128 + nt * 16 + l15] = (_Float16)hn;
    }
  }
}

// ---------------------------------------------------------------------------
// weight swizzle: fp32 [OUT x in_stride] -> f16 B-fragment order
// ---------------------------------------------------------------------------
__global__ void swizzle_kernel(const float* __restrict__ src, _Float16* __restrict__ dst,
                               int total, int KC, int col0, int in_stride) {
  int tid = blockIdx.x * blockDim.x + threadIdx.x;
  if (tid >= total) return;
  int e = tid & 15;
  int lane = (tid >> 4) & 31;
  int frag = tid >> 9;
  int kc = frag % KC;
  int nt = frag / KC;
  int n = nt * 16 + (lane & 15);
  int hi = lane >> 4;
  int k = kc * 32 + ((e >> 3) << 4) + hi * 8 + (e & 7);
  dst[tid] = (_Float16)src[(size_t)n * in_stride + col0 + k];
}

// ---------------------------------------------------------------------------
// Phase 1: encoder-1 (forward over 8 velocity steps)
// ---------------------------------------------------------------------------
__global__ __launch_bounds__(256) void enc1_kernel(
    const float* __restrict__ obs, const float* __restrict__ emb_w,
    const float* __restrict__ emb_b, const _Float16* __restrict__ wih,
    const _Float16* __restrict__ whh, const float* __restrict__ bih,
    const float* __restrict__ bhh, _Float16* __restrict__ h1_out) {
  __shared__ __attribute__((aligned(16))) _Float16 s_h[8 * 16 * 128];
  __shared__ float s_embw[128], s_embb[64], s_bias[512];
  const int tid = threadIdx.x;
  if (tid < 128) s_embw[tid] = emb_w[tid];
  if (tid < 64)  s_embb[tid] = emb_b[tid];
  s_bias[tid]       = bih[tid] + bhh[tid];
  s_bias[tid + 256] = bih[tid + 256] + bhh[tid + 256];
  __syncthreads();

  const int wave = tid >> 5, lane = tid & 31;
  const int hi = lane >> 4, l15 = lane & 15;
  const int abase = (blockIdx.x * 8 + wave) * 16;
  const int agent = abase + l15;
  _Float16* s_hw = s_h + wave * 16 * 128;

  v16h hfrag[4], xfrag[2];
  v8f c[8];
  {
    v16h z;
#pragma unroll
    for (int e = 0; e < 16; ++e) z[e] = (_Float16)0.0f;
#pragma unroll
    for (int q = 0; q < 4; ++q) hfrag[q] = z;
    v8f z8 = {0, 0, 0, 0, 0, 0, 0, 0};
#pragma unroll
    for (int nt = 0; nt < 8; ++nt) c[nt] = z8;
  }

  int lz = 0;
#pragma unroll 1
  for (int t = 0; t < 8; ++t) {
    asm volatile("" : "+s"(lz));  // defeat LICM -> no hoist/spill of weights
    float vx = obs[((size_t)(t + 1) * N_AGENTS + agent) * 2 + 0] -
               obs[((size_t)t * N_AGENTS + agent) * 2 + 0];
    float vy = obs[((size_t)(t + 1) * N_AGENTS + agent) * 2 + 1] -
               obs[((size_t)t * N_AGENTS + agent) * 2 + 1];
    emb_frags(vx, vy, s_embw, s_embb, hi, xfrag, lz);
    lstm_step<false>(wih, whh, nullptr, xfrag, hfrag, nullptr, s_bias, lane, c,
                     s_hw, lz);
    __syncthreads();
    load_frags_row(s_hw + l15 * 128, hi, hfrag);
    __syncthreads();
  }
  // final h1 (f16, row-major [agent][128])
#pragma unroll
  for (int j = 0; j < 8; ++j) {
    *(v8h*)(h1_out + (size_t)(abase + l15) * 128 + hi * 64 + j * 8) =
        *(const v8h*)(s_hw + l15 * 128 + hi * 64 + j * 8);
  }
}

// ---------------------------------------------------------------------------
// Phase 2: encoder-2 (reversed time, input = [emb(v), h1_final])
// ---------------------------------------------------------------------------
__global__ __launch_bounds__(256) void enc2_kernel(
    const float* __restrict__ obs, const float* __restrict__ emb_w,
    const float* __restrict__ emb_b, const _Float16* __restrict__ wih_x,
    const _Float16* __restrict__ wih_h, const _Float16* __restrict__ whh,
    const float* __restrict__ bih, const float* __restrict__ bhh,
    const _Float16* __restrict__ h1_in, _Float16* __restrict__ h2_out,
    float* __restrict__ c2_out) {
  __shared__ __attribute__((aligned(16))) _Float16 s_h[8 * 16 * 128];
  __shared__ float s_embw[128], s_embb[64], s_bias[512];
  const int tid = threadIdx.x;
  if (tid < 128) s_embw[tid] = emb_w[tid];
  if (tid < 64)  s_embb[tid] = emb_b[tid];
  s_bias[tid]       = bih[tid] + bhh[tid];
  s_bias[tid + 256] = bih[tid + 256] + bhh[tid + 256];
  __syncthreads();

  const int wave = tid >> 5, lane = tid & 31;
  const int hi = lane >> 4, l15 = lane & 15;
  const int abase = (blockIdx.x * 8 + wave) * 16;
  const int agent = abase + l15;
  _Float16* s_hw = s_h + wave * 16 * 128;

  v16h hfrag[4], xfrag[2], h1frag[4];
  v8f c[8];
  {
    v16h z;
#pragma unroll
    for (int e = 0; e < 16; ++e) z[e] = (_Float16)0.0f;
#pragma unroll
    for (int q = 0; q < 4; ++q) hfrag[q] = z;
    v8f z8 = {0, 0, 0, 0, 0, 0, 0, 0};
#pragma unroll
    for (int nt = 0; nt < 8; ++nt) c[nt] = z8;
  }
  // constant h1 A-fragments for this agent tile (persist in registers)
  load_frags_row(h1_in + (size_t)agent * 128, hi, h1frag);

  int lz = 0;
#pragma unroll 1
  for (int s = 0; s < 8; ++s) {
    asm volatile("" : "+s"(lz));  // defeat LICM -> no hoist/spill of weights
    int t = 7 - s;  // reversed velocities
    float vx = obs[((size_t)(t + 1) * N_AGENTS + agent) * 2 + 0] -
               obs[((size_t)t * N_AGENTS + agent) * 2 + 0];
    float vy = obs[((size_t)(t + 1) * N_AGENTS + agent) * 2 + 1] -
               obs[((size_t)t * N_AGENTS + agent) * 2 + 1];
    emb_frags(vx, vy, s_embw, s_embb, hi, xfrag, lz);
    lstm_step<true>(wih_x, whh, wih_h, xfrag, hfrag, h1frag, s_bias, lane, c,
                    s_hw, lz);
    __syncthreads();
    load_frags_row(s_hw + l15 * 128, hi, hfrag);
    __syncthreads();
  }
  // store h2 (f16 row-major) and c2 (f32, from D-layout registers)
#pragma unroll
  for (int j = 0; j < 8; ++j) {
    *(v8h*)(h2_out + (size_t)(abase + l15) * 128 + hi * 64 + j * 8) =
        *(const v8h*)(s_hw + l15 * 128 + hi * 64 + j * 8);
  }
#pragma unroll
  for (int nt = 0; nt < 8; ++nt) {
#pragma unroll
    for (int r = 0; r < 8; ++r) {
      c2_out[(size_t)(abase + r + hi * 8) * 128 + nt * 16 + l15] = c[nt][r];
    }
  }
}

// ---------------------------------------------------------------------------
// Phase 3: decoder (autoregressive rollout + hidden2normal head)
// ---------------------------------------------------------------------------
__global__ __launch_bounds__(256) void dec_kernel(
    const float* __restrict__ obs, const float* __restrict__ emb_w,
    const float* __restrict__ emb_b, const _Float16* __restrict__ wih,
    const _Float16* __restrict__ whh, const float* __restrict__ bih,
    const float* __restrict__ bhh, const _Float16* __restrict__ h2_in,
    const float* __restrict__ c2_in, const float* __restrict__ h2n_w,
    const float* __restrict__ h2n_b, const int* __restrict__ npred_p,
    float* __restrict__ out) {
  __shared__ __attribute__((aligned(16))) _Float16 s_h[8 * 16 * 128];
  __shared__ float s_embw[128], s_embb[64], s_bias[512];
  __shared__ float s_h2nw[5 * 128], s_h2nb[5];
  const int tid = threadIdx.x;
  if (tid < 128) s_embw[tid] = emb_w[tid];
  if (tid < 64)  s_embb[tid] = emb_b[tid];
  s_bias[tid]       = bih[tid] + bhh[tid];
  s_bias[tid + 256] = bih[tid + 256] + bhh[tid + 256];
  if (tid < 5 * 128) s_h2nw[tid] = h2n_w[tid];
  if (tid < 5)       s_h2nb[tid] = h2n_b[tid];
  __syncthreads();

  const int np = *npred_p;
  const int wave = tid >> 5, lane = tid & 31;
  const int hi = lane >> 4, l15 = lane & 15;
  const int abase = (blockIdx.x * 8 + wave) * 16;
  const int agent = abase + l15;
  _Float16* s_hw = s_h + wave * 16 * 128;

  v16h hfrag[4], xfrag[2];
  v8f c[8];
  load_frags_row(h2_in + (size_t)agent * 128, hi, hfrag);
#pragma unroll
  for (int nt = 0; nt < 8; ++nt) {
#pragma unroll
    for (int r = 0; r < 8; ++r) {
      c[nt][r] = c2_in[(size_t)(abase + r + hi * 8) * 128 + nt * 16 + l15];
    }
  }

  // positions: p_prev = observed[-2], p_curr = observed[-1]
  float ppx = obs[((size_t)7 * N_AGENTS + agent) * 2 + 0];
  float ppy = obs[((size_t)7 * N_AGENTS + agent) * 2 + 1];
  float pcx = obs[((size_t)8 * N_AGENTS + agent) * 2 + 0];
  float pcy = obs[((size_t)8 * N_AGENTS + agent) * 2 + 1];

  float* rel  = out;
  float* pred = out + (size_t)np * N_AGENTS * 5;

  int lz = 0;
#pragma unroll 1
  for (int s = 0; s < np; ++s) {
    asm volatile("" : "+s"(lz));  // defeat LICM -> no hoist/spill of weights
    float vx = pcx - ppx, vy = pcy - ppy;
    emb_frags(vx, vy, s_embw, s_embb, hi, xfrag, lz);
    lstm_step<false>(wih, whh, nullptr, xfrag, hfrag, nullptr, s_bias, lane, c,
                     s_hw, lz);
    __syncthreads();
    load_frags_row(s_hw + l15 * 128, hi, hfrag);
    __syncthreads();

    // hidden2normal: n = h @ W.T + b (5 outputs), half-wave reduce via shfl
    float pt[5];
#pragma unroll
    for (int o = 0; o < 5; ++o) pt[o] = 0.0f;
#pragma unroll
    for (int q = 0; q < 4; ++q) {
#pragma unroll
      for (int e = 0; e < 16; ++e) {
        int k = q * 32 + ((e >> 3) << 4) + hi * 8 + (e & 7);
        float hv = (float)hfrag[q][e];
#pragma unroll
        for (int o = 0; o < 5; ++o) pt[o] += hv * s_h2nw[o * 128 + k + lz];
      }
    }
    float n0 = pt[0] + __shfl_xor(pt[0], 16, 32) + s_h2nb[0];
    float n1 = pt[1] + __shfl_xor(pt[1], 16, 32) + s_h2nb[1];
    float n2 = pt[2] + __shfl_xor(pt[2], 16, 32) + s_h2nb[2];
    float n3 = pt[3] + __shfl_xor(pt[3], 16, 32) + s_h2nb[3];
    float n4 = pt[4] + __shfl_xor(pt[4], 16, 32) + s_h2nb[4];

    float pnx = pcx + n0, pny = pcy + n1;
    if (lane < 16) {
      size_t rb = ((size_t)s * N_AGENTS + agent) * 5;
      rel[rb + 0] = n0;
      rel[rb + 1] = n1;
      rel[rb + 2] = 0.01f + 0.2f * sigf(n2);
      rel[rb + 3] = 0.01f + 0.2f * sigf(n3);
      rel[rb + 4] = 0.7f * sigf(n4);
      size_t pb = ((size_t)s * N_AGENTS + agent) * 2;
      pred[pb + 0] = pnx;
      pred[pb + 1] = pny;
    }
    ppx = pcx; ppy = pcy; pcx = pnx; pcy = pny;
  }
}

// ---------------------------------------------------------------------------
// host entry
// ---------------------------------------------------------------------------
extern "C" void kernel_launch(void* const* d_in, const int* in_sizes, int n_in,
                              void* d_out, int out_size, void* d_ws, size_t ws_size,
                              hipStream_t stream) {
  (void)in_sizes; (void)n_in; (void)out_size; (void)ws_size;
  const float* obs    = (const float*)d_in[0];
  const int*   npred  = (const int*)d_in[3];
  const float* emb_w  = (const float*)d_in[4];
  const float* emb_b  = (const float*)d_in[5];
  const float* e1_ih  = (const float*)d_in[6];
  const float* e1_hh  = (const float*)d_in[7];
  const float* e1_bi  = (const float*)d_in[8];
  const float* e1_bh  = (const float*)d_in[9];
  const float* e2_ih  = (const float*)d_in[10];
  const float* e2_hh  = (const float*)d_in[11];
  const float* e2_bi  = (const float*)d_in[12];
  const float* e2_bh  = (const float*)d_in[13];
  const float* d_ih   = (const float*)d_in[14];
  const float* d_hh   = (const float*)d_in[15];
  const float* d_bi   = (const float*)d_in[16];
  const float* d_bh   = (const float*)d_in[17];
  const float* h2n_w  = (const float*)d_in[18];
  const float* h2n_b  = (const float*)d_in[19];

  char* ws = (char*)d_ws;
  _Float16* w_e1_ih  = (_Float16*)(ws + OFF_E1_IH);
  _Float16* w_e1_hh  = (_Float16*)(ws + OFF_E1_HH);
  _Float16* w_e2_ihx = (_Float16*)(ws + OFF_E2_IHX);
  _Float16* w_e2_ihh = (_Float16*)(ws + OFF_E2_IHH);
  _Float16* w_e2_hh  = (_Float16*)(ws + OFF_E2_HH);
  _Float16* w_d_ih   = (_Float16*)(ws + OFF_D_IH);
  _Float16* w_d_hh   = (_Float16*)(ws + OFF_D_HH);
  _Float16* h1       = (_Float16*)(ws + OFF_H1);
  _Float16* h2       = (_Float16*)(ws + OFF_H2);
  float*    c2       = (float*)   (ws + OFF_C2);

  auto swz = [&](const float* src, _Float16* dst, int total, int KC, int col0, int stride) {
    swizzle_kernel<<<(total + 255) / 256, 256, 0, stream>>>(src, dst, total, KC, col0, stride);
  };
  swz(e1_ih, w_e1_ih,  512 * 64,  2, 0,  64);
  swz(e1_hh, w_e1_hh,  512 * 128, 4, 0,  128);
  swz(e2_ih, w_e2_ihx, 512 * 64,  2, 0,  192);
  swz(e2_ih, w_e2_ihh, 512 * 128, 4, 64, 192);
  swz(e2_hh, w_e2_hh,  512 * 128, 4, 0,  128);
  swz(d_ih,  w_d_ih,   512 * 64,  2, 0,  64);
  swz(d_hh,  w_d_hh,   512 * 128, 4, 0,  128);

  const int blocks = N_AGENTS / 128;  // 512 blocks, 8 waves x 16 agents each
  enc1_kernel<<<blocks, 256, 0, stream>>>(obs, emb_w, emb_b, w_e1_ih, w_e1_hh,
                                          e1_bi, e1_bh, h1);
  enc2_kernel<<<blocks, 256, 0, stream>>>(obs, emb_w, emb_b, w_e2_ihx, w_e2_ihh,
                                          w_e2_hh, e2_bi, e2_bh, h1, h2, c2);
  dec_kernel<<<blocks, 256, 0, stream>>>(obs, emb_w, emb_b, w_d_ih, w_d_hh,
                                         d_bi, d_bh, h2, c2, h2n_w, h2n_b,
                                         npred, (float*)d_out);
}
